// ScaledDotProductAttention_2869038154032
// MI455X (gfx1250) — matrix operand order repaired and checked by reading, not compiled, not run on hardware
//
#include <hip/hip_runtime.h>
#include <hip/hip_bf16.h>

// MI455X (gfx1250) attention:
//  - S^T = K·Q^T via v_wmma_f32_16x16x4_f32 (exact fp32 scores, K-dim = D = 64)
//  - P·V via v_wmma_f32_16x16x32_bf16 (K-dim = L = 2048 dominates FLOPs; bf16 8x/instr)
//  - cross score GEMM via v_wmma_f32_16x16x4_f32 (exact, output compared directly)
//  - flash online softmax, 8 kv-segments per q-tile merged with LDS + ds_add_f32
//  - nontemporal stores for the 268MB score stream (L2 is 192MB)

typedef float  v2f   __attribute__((ext_vector_type(2)));
typedef float  v4f   __attribute__((ext_vector_type(4)));
typedef float  v8f   __attribute__((ext_vector_type(8)));
typedef __bf16 v16bf __attribute__((ext_vector_type(16)));

#define B_ 16
#define L_ 2048
#define D_ 64
#define SCALE 0.125f   // 1/sqrt(64)
#define NEG_INF (-__builtin_inff())

static __device__ inline v8f v8f_zero() {
  v8f z;
#pragma unroll
  for (int i = 0; i < 8; ++i) z[i] = 0.0f;
  return z;
}

// ---------------------------------------------------------------------------
// Attention kernel: one block = one (batch, 16-row q-tile); 8 waves split kv.
// ---------------------------------------------------------------------------
__global__ __launch_bounds__(256) void attn_kernel(
    const float* __restrict__ q, const float* __restrict__ k,
    const float* __restrict__ v, const unsigned char* __restrict__ mask,
    float* __restrict__ ctx)
{
  __shared__ __align__(16) float Om[16][64];   // block-level context accumulator
  __shared__ float redm[8][16];                // per-wave running max, per q
  __shared__ float reds[8][16];                // per-wave running sum, per q

  const int tid  = threadIdx.x;
  const int wave = tid >> 5;
  const int lane = tid & 31;
  const int h    = lane >> 4;   // lane half (0/1)
  const int ln   = lane & 15;

  const int b  = blockIdx.x / (L_ / 16);
  const int q0 = (blockIdx.x % (L_ / 16)) * 16;

  for (int i = tid; i < 16 * 64; i += 256) ((float*)Om)[i] = 0.0f;
  __syncthreads();

  // Hoisted Q B-fragments for f32 WMMA: lane holds col n=q0+ln, K pair 4s+2h+{0,1}
  const float* qrow = q + ((size_t)b * L_ + q0 + ln) * D_;
  v2f qb[16];
#pragma unroll
  for (int s = 0; s < 16; ++s) qb[s] = *(const v2f*)(qrow + 4 * s + 2 * h);

  v8f O[4];
#pragma unroll
  for (int t = 0; t < 4; ++t) O[t] = v8f_zero();
  float m_run = NEG_INF;
  float s_run = 0.0f;

  const unsigned char* mrow = mask + ((size_t)b * L_ + q0 + ln) * (size_t)L_;
  const int kvbeg = wave * (L_ / 8);
  const int kvend = kvbeg + (L_ / 8);

  for (int kv0 = kvbeg; kv0 < kvend; kv0 += 32) {
    // --- S^T tiles: M = kv (16 each), N = q, K-dim = d (64 -> 16 steps of 4)
    const float* k0row = k + ((size_t)b * L_ + kv0 + ln) * D_;
    const float* k1row = k0row + 16 * D_;
    v8f S0 = v8f_zero();
    v8f S1 = v8f_zero();
#pragma unroll
    for (int s = 0; s < 16; ++s) {
      v2f ka = *(const v2f*)(k0row + 4 * s + 2 * h);
      S0 = __builtin_amdgcn_wmma_f32_16x16x4_f32(false, ka, false, qb[s],
                                                 (short)0, S0, false, false);
    }
#pragma unroll
    for (int s = 0; s < 16; ++s) {
      v2f ka = *(const v2f*)(k1row + 4 * s + 2 * h);
      S1 = __builtin_amdgcn_wmma_f32_16x16x4_f32(false, ka, false, qb[s],
                                                 (short)0, S1, false, false);
    }

    // --- scale + mask (mask==true -> -inf). Element (lane,r): q=q0+ln, kv=base+8h+r
    unsigned long long mb0 = *(const unsigned long long*)(mrow + kv0 + 8 * h);
    unsigned long long mb1 = *(const unsigned long long*)(mrow + kv0 + 16 + 8 * h);
    float tmax = NEG_INF;
#pragma unroll
    for (int r = 0; r < 8; ++r) {
      float a = S0[r] * SCALE;
      float c = S1[r] * SCALE;
      if ((mb0 >> (8 * r)) & 0xffull) a = NEG_INF;
      if ((mb1 >> (8 * r)) & 0xffull) c = NEG_INF;
      S0[r] = a; S1[r] = c;
      tmax = fmaxf(tmax, fmaxf(a, c));
    }
    tmax = fmaxf(tmax, __shfl_xor(tmax, 16, 32));   // combine kv-halves (same q)
    float m_new = fmaxf(m_run, tmax);
    float alpha = __expf(m_run - m_new);

    float csum = 0.0f;
#pragma unroll
    for (int r = 0; r < 8; ++r) {
      float p0 = __expf(S0[r] - m_new);
      float p1 = __expf(S1[r] - m_new);
      S0[r] = p0; S1[r] = p1;
      csum += p0 + p1;
    }
    csum += __shfl_xor(csum, 16, 32);
    s_run = s_run * alpha + csum;
    m_run = m_new;

    // --- rescale O: row q' = r + 8h in O-layout; alpha for q' lives in lane q'
#pragma unroll
    for (int r = 0; r < 8; ++r) {
      float ar = __shfl(alpha, r + 8 * h, 32);
#pragma unroll
      for (int t = 0; t < 4; ++t) O[t][r] *= ar;
    }

    // --- P as bf16 A-fragment: lane-local identity with S^T C-layout (r == e)
    v16bf pa;
#pragma unroll
    for (int e = 0; e < 8; ++e) {
      pa[e]     = (__bf16)S0[e];
      pa[8 + e] = (__bf16)S1[e];
    }

    // --- V B-fragments (K=kv, N=d): lane needs V[kv0+8h+e+(e&8)][16t+ln]
    const float* vbase = v + ((size_t)b * L_ + kv0 + 8 * h) * D_ + ln;
#pragma unroll
    for (int t = 0; t < 4; ++t) {
      v16bf vb;
#pragma unroll
      for (int e = 0; e < 16; ++e) vb[e] = (__bf16)vbase[(size_t)(e + (e & 8)) * D_ + 16 * t];
      O[t] = __builtin_amdgcn_wmma_f32_16x16x32_bf16(false, pa, false, vb,
                                                     (short)0, O[t], false, false);
    }
  }

  // --- merge the 8 kv-segments: log-sum-exp over waves
  if (lane < 16) { redm[wave][ln] = m_run; reds[wave][ln] = s_run; }
  __syncthreads();
  float M = NEG_INF;
#pragma unroll
  for (int w = 0; w < 8; ++w) M = fmaxf(M, redm[w][ln]);
  float Ssum = 0.0f;
#pragma unroll
  for (int w = 0; w < 8; ++w) Ssum += reds[w][ln] * __expf(redm[w][ln] - M);
  float f = __expf(m_run - M) / Ssum;

#pragma unroll
  for (int r = 0; r < 8; ++r) {
    float fr = __shfl(f, r + 8 * h, 32);
#pragma unroll
    for (int t = 0; t < 4; ++t) O[t][r] *= fr;
  }

  // --- accumulate segments into LDS (ds_add_f32)
#pragma unroll
  for (int t = 0; t < 4; ++t)
#pragma unroll
    for (int r = 0; r < 8; ++r)
      atomicAdd(&Om[r + 8 * h][16 * t + ln], O[t][r]);
  __syncthreads();

  // --- cooperative store of the 16x64 context tile
  float* crow = ctx + ((size_t)b * L_ + q0) * D_;
  for (int i = tid; i < (16 * 64) / 4; i += 256) {
    v4f val = ((const v4f*)Om)[i];
    __builtin_nontemporal_store(val, (v4f*)crow + i);
  }
}

// ---------------------------------------------------------------------------
// Cross-score GEMM: out[b][i][j] = dot(cross1[b][i], cross2[b][j]), exact f32.
// One wave per 16x16 output tile; 8 waves per block.
// ---------------------------------------------------------------------------
__global__ __launch_bounds__(256) void cross_kernel(
    const float* __restrict__ c1, const float* __restrict__ c2,
    float* __restrict__ out)
{
  const int tid  = threadIdx.x;
  const int wave = tid >> 5;
  const int lane = tid & 31;
  const int h    = lane >> 4;
  const int ln   = lane & 15;

  const int id  = blockIdx.x * 8 + wave;
  const int b   = id >> 14;          // / (128*128)
  const int rem = id & 16383;
  const int it  = rem >> 7;
  const int jt  = rem & 127;

  const float* arow = c1 + ((size_t)b * L_ + it * 16 + ln) * D_;
  const float* brow = c2 + ((size_t)b * L_ + jt * 16 + ln) * D_;

  v8f acc = v8f_zero();
#pragma unroll
  for (int s = 0; s < 16; ++s) {
    v2f af = *(const v2f*)(arow + 4 * s + 2 * h);   // A: M=i, K=d
    v2f bf = *(const v2f*)(brow + 4 * s + 2 * h);   // B: N=j, K=d (from c2 rows)
    acc = __builtin_amdgcn_wmma_f32_16x16x4_f32(false, af, false, bf,
                                                (short)0, acc, false, false);
  }

  float* orow = out + (size_t)b * L_ * L_ + (size_t)(it * 16) * L_ + jt * 16 + ln;
#pragma unroll
  for (int r = 0; r < 8; ++r)
    __builtin_nontemporal_store(acc[r], orow + (size_t)(r + 8 * h) * L_);
}

// ---------------------------------------------------------------------------
extern "C" void kernel_launch(void* const* d_in, const int* in_sizes, int n_in,
                              void* d_out, int out_size, void* d_ws, size_t ws_size,
                              hipStream_t stream) {
  const float*         q    = (const float*)d_in[0];
  const float*         k    = (const float*)d_in[1];
  const float*         v    = (const float*)d_in[2];
  const float*         c1   = (const float*)d_in[3];
  const float*         c2   = (const float*)d_in[4];
  const unsigned char* mask = (const unsigned char*)d_in[5];

  float* ctx   = (float*)d_out;
  float* score = (float*)d_out + (size_t)B_ * L_ * D_;

  attn_kernel<<<B_ * (L_ / 16), 256, 0, stream>>>(q, k, v, mask, ctx);
  cross_kernel<<<(B_ * (L_ / 16) * (L_ / 16)) / 8, 256, 0, stream>>>(c1, c2, score);
}
